// Attention_11579231830783
// MI455X (gfx1250) — matrix operand order repaired and checked by reading, not compiled, not run on hardware
//
#include <hip/hip_runtime.h>

// ---------------------------------------------------------------------------
// Types for gfx1250 WMMA (wave32): v_wmma_f32_16x16x32_bf16
// ---------------------------------------------------------------------------
typedef __bf16 v8bf  __attribute__((ext_vector_type(8)));
typedef __bf16 v16bf __attribute__((ext_vector_type(16)));
typedef float  v8f   __attribute__((ext_vector_type(8)));
typedef float  v4f   __attribute__((ext_vector_type(4)));

#define DIMC   512
#define NHEAD  8
#define HDIM   64
#define NTOK   4096
#define NBATCH 2
#define NROWS  (NBATCH * NTOK)   // 8192
#define QKV_N  (3 * DIMC)        // 1536
#define ATTN_SCALE 0.125f        // 64^-0.5

// ---------------------------------------------------------------------------
// Helpers
// ---------------------------------------------------------------------------
__device__ __forceinline__ v8f zero8() {
  v8f v;
#pragma unroll
  for (int i = 0; i < 8; ++i) v[i] = 0.0f;
  return v;
}

__device__ __forceinline__ v8f wmma_bf16(v16bf a, v16bf b, v8f c) {
  // D = A(16x32 bf16) * B(32x16 bf16) + C(16x16 f32)
  return __builtin_amdgcn_wmma_f32_16x16x32_bf16(
      /*neg_a=*/false, a, /*neg_b=*/false, b,
      /*c_mod=*/(short)0, c, /*reuse_a=*/false, /*reuse_b=*/false);
}

// A-fragment (16x32, MxK): lane L holds row M=L%16, K chunks {g*8..g*8+7, g*8+16..g*8+23}
__device__ __forceinline__ v16bf load_frag_a(const __bf16* __restrict__ base, int stride) {
  const int lane = threadIdx.x & 31;
  const __bf16* p = base + (lane & 15) * stride + ((lane >> 4) << 3);
  v8bf lo = *(const v8bf*)p;
  v8bf hi = *(const v8bf*)(p + 16);
  v16bf r;
#pragma unroll
  for (int i = 0; i < 8; ++i) { r[i] = lo[i]; r[i + 8] = hi[i]; }
  return r;
}

// B-fragment (32x16, KxN): lane L holds column N=L%16, K chunks {g*8..g*8+7, g*8+16..g*8+23}.
// Source rows in memory are the N columns (row n = B column n), contiguous in K.
__device__ __forceinline__ v16bf load_frag_b(const __bf16* __restrict__ base, int stride) {
  const int lane = threadIdx.x & 31;
  const __bf16* p = base + (lane & 15) * stride + ((lane >> 4) << 3);
  v8bf lo = *(const v8bf*)p;
  v8bf hi = *(const v8bf*)(p + 16);
  v16bf r;
#pragma unroll
  for (int i = 0; i < 8; ++i) { r[i] = lo[i]; r[i + 8] = hi[i]; }
  return r;
}

// Butterfly reductions within a 16-lane group (xor masks 1..8 stay inside group)
__device__ __forceinline__ float rmax16(float v) {
#pragma unroll
  for (int m = 8; m >= 1; m >>= 1) v = fmaxf(v, __shfl_xor(v, m, 32));
  return v;
}
__device__ __forceinline__ float rsum16(float v) {
#pragma unroll
  for (int m = 8; m >= 1; m >>= 1) v += __shfl_xor(v, m, 32);
  return v;
}

// ---------------------------------------------------------------------------
// Shared GEMM core: one wave computes a 32x64 f32 tile of A(bf16) x B^T(bf16).
// Fully unrolled k-loop (K=512, step 32) with register double-buffering so
// next-step global_load_b128s are in flight during the 8 WMMAs of this step.
// ---------------------------------------------------------------------------
__device__ __forceinline__ void gemm_core_32x64(
    const __bf16* __restrict__ aBase,   // points at row0 * DIMC
    const __bf16* __restrict__ bBase,   // points at (col0) * DIMC (rows = out channels)
    v8f acc[2][4]) {
  v16bf a[2], b[4];
#pragma unroll
  for (int mi = 0; mi < 2; ++mi) a[mi] = load_frag_a(aBase + mi * 16 * DIMC, DIMC);
#pragma unroll
  for (int t = 0; t < 4; ++t) b[t] = load_frag_b(bBase + t * 16 * DIMC, DIMC);

#pragma unroll
  for (int it = 0; it < 15; ++it) {
    const int kn = (it + 1) * 32;
    v16bf an[2], bn[4];
#pragma unroll
    for (int mi = 0; mi < 2; ++mi)
      an[mi] = load_frag_a(aBase + mi * 16 * DIMC + kn, DIMC);
#pragma unroll
    for (int t = 0; t < 4; ++t)
      bn[t] = load_frag_b(bBase + t * 16 * DIMC + kn, DIMC);
#pragma unroll
    for (int mi = 0; mi < 2; ++mi)
#pragma unroll
      for (int t = 0; t < 4; ++t)
        acc[mi][t] = wmma_bf16(a[mi], b[t], acc[mi][t]);
#pragma unroll
    for (int mi = 0; mi < 2; ++mi) a[mi] = an[mi];
#pragma unroll
    for (int t = 0; t < 4; ++t) b[t] = bn[t];
  }
  // tail step (no prefetch)
#pragma unroll
  for (int mi = 0; mi < 2; ++mi)
#pragma unroll
    for (int t = 0; t < 4; ++t)
      acc[mi][t] = wmma_bf16(a[mi], b[t], acc[mi][t]);
}

// ---------------------------------------------------------------------------
// fp32 -> bf16 conversion (vectorized x4)
// ---------------------------------------------------------------------------
__global__ __launch_bounds__(256) void cvt_f32_bf16(const float* __restrict__ in,
                                                    __bf16* __restrict__ out, int n4) {
  int i = blockIdx.x * 256 + threadIdx.x;
  if (i < n4) {
    v4f v = *(const v4f*)(in + i * 4);
    __bf16* o = out + i * 4;
    o[0] = (__bf16)v[0]; o[1] = (__bf16)v[1]; o[2] = (__bf16)v[2]; o[3] = (__bf16)v[3];
  }
}

// ---------------------------------------------------------------------------
// QKV GEMM: [8192,512](bf16) x W^T(1536x512 rows, bf16) + b -> Q,K (B,H,N,Dh), V^T (B,H,Dh,N)
// One wave computes a 32x64 output tile. 6144 waves, 128-thread blocks.
// ---------------------------------------------------------------------------
__global__ __launch_bounds__(128) void qkv_gemm_kernel(
    const __bf16* __restrict__ xb, const __bf16* __restrict__ wb,
    const float* __restrict__ bias,
    __bf16* __restrict__ Q, __bf16* __restrict__ K, __bf16* __restrict__ Vt) {
  const int gw = blockIdx.x * 4 + (threadIdx.x >> 5);
  const int lane = threadIdx.x & 31;
  const int mt = gw / 24;              // 0..255 : 32-row tile
  const int ns = gw % 24;              // 0..23  : 64-output strip
  const int row0 = mt * 32;

  v8f acc[2][4];
#pragma unroll
  for (int mi = 0; mi < 2; ++mi)
#pragma unroll
    for (int t = 0; t < 4; ++t) acc[mi][t] = zero8();

  gemm_core_32x64(xb + row0 * DIMC, wb + (ns * 64) * DIMC, acc);

  const int g = lane >> 4, cl = lane & 15;
#pragma unroll
  for (int t = 0; t < 4; ++t) {
    const int o = ns * 64 + t * 16 + cl;    // output channel 0..1535
    const float bs = bias[o];
    const int sec = o >> 9;                 // 0=Q 1=K 2=V
    const int oc = o & 511;
    const int h = oc >> 6, d = oc & 63;
#pragma unroll
    for (int mi = 0; mi < 2; ++mi) {
#pragma unroll
      for (int r = 0; r < 8; ++r) {
        const int rr = row0 + mi * 16 + r + 8 * g;  // global row 0..8191
        const int bi = rr >> 12, tok = rr & 4095;
        const float v = acc[mi][t][r] + bs;
        const int bh = bi * NHEAD + h;
        if (sec == 0) {
          Q[((bh * NTOK + tok) << 6) + d] = (__bf16)(v * ATTN_SCALE);
        } else if (sec == 1) {
          K[((bh * NTOK + tok) << 6) + d] = (__bf16)v;
        } else {
          Vt[((bh << 6) + d) * NTOK + tok] = (__bf16)v;
        }
      }
    }
  }
}

// ---------------------------------------------------------------------------
// Flash attention: one wave handles 32 queries of one (b,h); keys in chunks of 32.
// Per chunk: 8 score WMMAs + 8 PV WMMAs; online softmax via shfl butterflies.
// V fragments are loaded together with K fragments at the top of each chunk so
// the score WMMAs + softmax VALU overlap the V loads (partial loadcnt waits).
// P transposed (C/D layout -> A-frag layout) through 2KB of per-wave LDS.
// ---------------------------------------------------------------------------
__global__ __launch_bounds__(32) void attn_kernel(
    const __bf16* __restrict__ Q, const __bf16* __restrict__ K,
    const __bf16* __restrict__ Vt, __bf16* __restrict__ O) {
  __shared__ __bf16 pbuf[2][16 * 32];

  const int lane = threadIdx.x & 31;
  const int bid = blockIdx.x;
  const int qb = bid & 127;
  const int h  = (bid >> 7) & 7;
  const int b  = bid >> 10;
  const int bh = b * NHEAD + h;
  const __bf16* Qbh = Q  + (size_t)bh * NTOK * HDIM;
  const __bf16* Kbh = K  + (size_t)bh * NTOK * HDIM;
  const __bf16* Vbh = Vt + (size_t)bh * HDIM * NTOK;
  const int q0 = qb * 32;
  const int g = lane >> 4, cl = lane & 15;

  // Q fragments (persist): 2 q-tiles x 2 d-chunks
  v16bf qf[2][2];
#pragma unroll
  for (int qt = 0; qt < 2; ++qt)
#pragma unroll
    for (int c = 0; c < 2; ++c)
      qf[qt][c] = load_frag_a(Qbh + (q0 + qt * 16) * HDIM + c * 32, HDIM);

  v8f acc[2][4];
  float mrow[2][8], lrow[2][8];
#pragma unroll
  for (int qt = 0; qt < 2; ++qt) {
#pragma unroll
    for (int dt = 0; dt < 4; ++dt) acc[qt][dt] = zero8();
#pragma unroll
    for (int r = 0; r < 8; ++r) { mrow[qt][r] = -1e30f; lrow[qt][r] = 0.0f; }
  }

  for (int kc = 0; kc < NTOK; kc += 32) {
    // K fragments (needed first) then V fragments (needed last) — issue all
    // 16 b128 loads up front; scores wait only for the K portion.
    v16bf kf[2][2];
#pragma unroll
    for (int kt = 0; kt < 2; ++kt)
#pragma unroll
      for (int c = 0; c < 2; ++c)
        kf[kt][c] = load_frag_b(Kbh + (kc + kt * 16) * HDIM + c * 32, HDIM);

    v16bf vf[4];
#pragma unroll
    for (int dt = 0; dt < 4; ++dt)
      vf[dt] = load_frag_b(Vbh + (dt * 16) * NTOK + kc, NTOK);

#pragma unroll
    for (int qt = 0; qt < 2; ++qt) {
      v8f s0 = zero8(), s1 = zero8();
      s0 = wmma_bf16(qf[qt][0], kf[0][0], s0);
      s0 = wmma_bf16(qf[qt][1], kf[0][1], s0);
      s1 = wmma_bf16(qf[qt][0], kf[1][0], s1);
      s1 = wmma_bf16(qf[qt][1], kf[1][1], s1);

      float alpha[8];
#pragma unroll
      for (int r = 0; r < 8; ++r) {
        float t = rmax16(fmaxf(s0[r], s1[r]));
        const float mn = fmaxf(mrow[qt][r], t);
        alpha[r] = __expf(mrow[qt][r] - mn);
        mrow[qt][r] = mn;
        const float p0 = __expf(s0[r] - mn);
        const float p1 = __expf(s1[r] - mn);
        lrow[qt][r] = lrow[qt][r] * alpha[r] + rsum16(p0 + p1);
        const int m = r + 8 * g;
        pbuf[qt][m * 32 + cl]      = (__bf16)p0;   // keys kc..kc+15
        pbuf[qt][m * 32 + 16 + cl] = (__bf16)p1;   // keys kc+16..kc+31
      }
#pragma unroll
      for (int dt = 0; dt < 4; ++dt)
#pragma unroll
        for (int r = 0; r < 8; ++r) acc[qt][dt][r] *= alpha[r];
    }

    // P A-fragments from LDS (same-wave DS ops are in-order; no barrier needed)
#pragma unroll
    for (int qt = 0; qt < 2; ++qt) {
      const __bf16* p = &pbuf[qt][(lane & 15) * 32 + ((lane >> 4) << 3)];
      v8bf lo = *(const v8bf*)p;
      v8bf hi = *(const v8bf*)(p + 16);
      v16bf pf;
#pragma unroll
      for (int i = 0; i < 8; ++i) { pf[i] = lo[i]; pf[i + 8] = hi[i]; }
#pragma unroll
      for (int dt = 0; dt < 4; ++dt) acc[qt][dt] = wmma_bf16(pf, vf[dt], acc[qt][dt]);
    }
  }

  // Epilogue: O[b, tok, h*64 + d] = acc / l  (bf16, token-major for proj GEMM)
#pragma unroll
  for (int qt = 0; qt < 2; ++qt) {
#pragma unroll
    for (int r = 0; r < 8; ++r) {
      const float inv = 1.0f / lrow[qt][r];
      const int tok = q0 + qt * 16 + r + 8 * g;
#pragma unroll
      for (int dt = 0; dt < 4; ++dt) {
        const int ch = h * HDIM + dt * 16 + cl;
        O[((size_t)b * NTOK + tok) * DIMC + ch] = (__bf16)(acc[qt][dt][r] * inv);
      }
    }
  }
}

// ---------------------------------------------------------------------------
// Projection GEMM: attn[8192,512](bf16) x proj_w^T + b -> out f32 [8192,512]
// One wave computes a 32x64 tile. 2048 waves, 128-thread blocks.
// ---------------------------------------------------------------------------
__global__ __launch_bounds__(128) void proj_gemm_kernel(
    const __bf16* __restrict__ ab, const __bf16* __restrict__ wb,
    const float* __restrict__ bias, float* __restrict__ out) {
  const int gw = blockIdx.x * 4 + (threadIdx.x >> 5);
  const int lane = threadIdx.x & 31;
  const int mt = gw >> 3;              // 0..255
  const int ns = gw & 7;               // 0..7 : 64-output strip
  const int row0 = mt * 32;

  v8f acc[2][4];
#pragma unroll
  for (int mi = 0; mi < 2; ++mi)
#pragma unroll
    for (int t = 0; t < 4; ++t) acc[mi][t] = zero8();

  gemm_core_32x64(ab + row0 * DIMC, wb + (ns * 64) * DIMC, acc);

  const int g = lane >> 4, cl = lane & 15;
#pragma unroll
  for (int t = 0; t < 4; ++t) {
    const int o = ns * 64 + t * 16 + cl;
    const float bs = bias[o];
#pragma unroll
    for (int mi = 0; mi < 2; ++mi) {
#pragma unroll
      for (int r = 0; r < 8; ++r) {
        const int rr = row0 + mi * 16 + r + 8 * g;
        out[rr * DIMC + o] = acc[mi][t][r] + bs;
      }
    }
  }
}

// ---------------------------------------------------------------------------
// Launch
// ---------------------------------------------------------------------------
extern "C" void kernel_launch(void* const* d_in, const int* in_sizes, int n_in,
                              void* d_out, int out_size, void* d_ws, size_t ws_size,
                              hipStream_t stream) {
  const float* x      = (const float*)d_in[0];   // [2,4096,512]
  const float* qkv_w  = (const float*)d_in[1];   // [1536,512]
  const float* qkv_b  = (const float*)d_in[2];   // [1536]
  const float* proj_w = (const float*)d_in[3];   // [512,512]
  const float* proj_b = (const float*)d_in[4];   // [512]
  float* out = (float*)d_out;

  const size_t N_X   = (size_t)NROWS * DIMC;     // 4194304
  const size_t N_WQ  = (size_t)QKV_N * DIMC;     // 786432
  const size_t N_WP  = (size_t)DIMC * DIMC;      // 262144
  const size_t N_QKV = (size_t)NBATCH * NHEAD * NTOK * HDIM;  // 4194304

  __bf16* xb    = (__bf16*)d_ws;
  __bf16* wqkv  = xb + N_X;
  __bf16* wproj = wqkv + N_WQ;
  __bf16* Qb    = wproj + N_WP;
  __bf16* Kb    = Qb + N_QKV;
  __bf16* Vtb   = Kb + N_QKV;
  __bf16* attn  = Vtb + N_QKV;

  cvt_f32_bf16<<<(int)(N_X  / 4 / 256), 256, 0, stream>>>(x, xb, (int)(N_X / 4));
  cvt_f32_bf16<<<(int)(N_WQ / 4 / 256), 256, 0, stream>>>(qkv_w, wqkv, (int)(N_WQ / 4));
  cvt_f32_bf16<<<(int)(N_WP / 4 / 256), 256, 0, stream>>>(proj_w, wproj, (int)(N_WP / 4));

  // 256 M-tiles * 24 strips = 6144 waves / 4 per block
  qkv_gemm_kernel<<<1536, 128, 0, stream>>>(xb, wqkv, qkv_b, Qb, Kb, Vtb);

  // 2 * 8 * (4096/32) = 2048 waves, one per block
  attn_kernel<<<2048, 32, 0, stream>>>(Qb, Kb, Vtb, attn);

  // 256 M-tiles * 8 strips = 2048 waves / 4 per block
  proj_gemm_kernel<<<512, 128, 0, stream>>>(attn, wproj, proj_b, out);
}
